// DualHeadCorrectorMutator_13357348290791
// MI455X (gfx1250) — matrix operand order repaired and checked by reading, not compiled, not run on hardware
//
#include <hip/hip_runtime.h>
#include <hip/hip_bf16.h>
#include <math.h>

typedef __attribute__((ext_vector_type(16))) _Float16 v16h;
typedef __attribute__((ext_vector_type(8)))  float    v8f;
typedef __attribute__((ext_vector_type(4)))  float    f4;

constexpr int CS   = 64;
constexpr int HID  = 64;
constexpr int NTOT = 33554432;
constexpr int NC   = NTOT / CS;      // 524288 chunks
constexpr int TILES = NC / 16;       // 32768 wave-tiles (16 chunks each)
constexpr int WPB  = 4;              // waves per block
constexpr int LSTR = 68;             // LDS row stride (floats) to dodge bank conflicts

// Pre-packed weight fragments: 26 frags x 32 lanes x 16 halves (f16)
constexpr int NFRAG = 26;
constexpr size_t WS_BYTES = (size_t)NFRAG * 32 * 16 * sizeof(_Float16);  // 26624 B

// fragment ids
constexpr int F_WE0 = 0, F_WE1 = 1;   // encoder, two K-halves
constexpr int F_BG  = 2;              // gate layer1: 2+2t / 3+2t, t=0..3
constexpr int F_BC  = 10;             // corr layer1: 10+2t / 11+2t (K=48 zero-padded)
constexpr int F_BD  = 18;             // corr layer2: 18+2t / 19+2t

// ---------------- math helpers (branch-free, hw trans ops) ----------------

__device__ __forceinline__ float fast_tanh(float x) {
#if __has_builtin(__builtin_amdgcn_tanhf)
  return __builtin_amdgcn_tanhf(x);        // v_tanh_f32 (CDNA5 TRANS op)
#else
  const float ax = __builtin_fabsf(x);
  const float e  = __expf(-2.0f * ax);
  const float r  = 1.0f - 2.0f * e * __builtin_amdgcn_rcpf(1.0f + e);
  return __builtin_copysignf(r, x);
#endif
}

__device__ __forceinline__ float fast_sigmoid(float x) {
  return __builtin_amdgcn_rcpf(1.0f + __expf(-x));   // v_exp + v_rcp
}

// ---------------- fragment helpers ----------------

__device__ __forceinline__ v8f wmma_f16(v16h a, v16h b, v8f c) {
  return __builtin_amdgcn_wmma_f32_16x16x32_f16(false, a, false, b, (short)0, c,
                                                false, false);
}

__device__ __forceinline__ v16h tohalf(const float* f) {
  v16h h;
#pragma unroll
  for (int i = 0; i < 16; ++i) h[i] = (_Float16)f[i];
  return h;
}

__device__ __forceinline__ v16h zero16h() {
  v16h h;
#pragma unroll
  for (int i = 0; i < 16; ++i) h[i] = (_Float16)0.0f;
  return h;
}

// bias broadcast as WMMA C operand (bias is per-column == per-lane in C layout)
__device__ __forceinline__ v8f bias8(float b) {
  v8f c;
#pragma unroll
  for (int i = 0; i < 8; ++i) c[i] = b;
  return c;
}

// A-fragment per-lane data: cols [k..k+7] and [k+16..k+23] of one row (LDS).
__device__ __forceinline__ void read_octets(const float* p, float* o) {
  const f4* q = reinterpret_cast<const f4*>(p);
  f4 x0 = q[0], x1 = q[1], x2 = q[4], x3 = q[5];
#pragma unroll
  for (int i = 0; i < 4; ++i) {
    o[i] = x0[i]; o[4 + i] = x1[i]; o[8 + i] = x2[i]; o[12 + i] = x3[i];
  }
}

// Same, but non-temporal (streaming 128MB global inputs).
__device__ __forceinline__ void read_octets_nt(const float* p, float* o) {
  const f4* q = reinterpret_cast<const f4*>(p);
  f4 x0 = __builtin_nontemporal_load(q + 0);
  f4 x1 = __builtin_nontemporal_load(q + 1);
  f4 x2 = __builtin_nontemporal_load(q + 4);
  f4 x3 = __builtin_nontemporal_load(q + 5);
#pragma unroll
  for (int i = 0; i < 4; ++i) {
    o[i] = x0[i]; o[4 + i] = x1[i]; o[8 + i] = x2[i]; o[12 + i] = x3[i];
  }
}

__device__ __forceinline__ void write_octets(float* p, const float* o) {
  f4 x0, x1, x2, x3;
#pragma unroll
  for (int i = 0; i < 4; ++i) {
    x0[i] = o[i]; x1[i] = o[4 + i]; x2[i] = o[8 + i]; x3[i] = o[12 + i];
  }
  f4* q = reinterpret_cast<f4*>(p);
  q[0] = x0; q[1] = x1; q[4] = x2; q[5] = x3;
}

// B-fragment from f32 weight matrix (prepack only): K [kbase..+7] and [kbase+16..+23] of one row.
__device__ __forceinline__ v16h load_bfrag(const float* wmat, int stride, int nrow,
                                           int kbase) {
  const f4* q = reinterpret_cast<const f4*>(wmat + nrow * stride + kbase);
  f4 x0 = q[0], x1 = q[1], x2 = q[4], x3 = q[5];
  float t[16];
#pragma unroll
  for (int i = 0; i < 4; ++i) {
    t[i] = x0[i]; t[4 + i] = x1[i]; t[8 + i] = x2[i]; t[12 + i] = x3[i];
  }
  return tohalf(t);
}

// Pre-packed B-fragment fetch: one aligned 32-byte f16 load, no conversion.
__device__ __forceinline__ v16h load_wsfrag(const _Float16* ws, int f, int lane) {
  return *reinterpret_cast<const v16h*>(ws + ((size_t)f * 32 + lane) * 16);
}

// ---------------- weight prepack kernel (runs once per launch, 1 wave) ----------------

__global__ void prepack_weights_kernel(const float* __restrict__ w_enc,
                                       const float* __restrict__ w_g1,
                                       const float* __restrict__ w_c1,
                                       const float* __restrict__ w_c2,
                                       _Float16* __restrict__ ws) {
  const int lane = threadIdx.x & 31;
  const int s = lane >> 4;
  const int n = lane & 15;
  const int kb = s * 8;
  auto put = [&](int f, v16h v) {
    *reinterpret_cast<v16h*>(ws + ((size_t)f * 32 + lane) * 16) = v;
  };
  put(F_WE0, load_bfrag(w_enc, CS, n, kb));
  put(F_WE1, load_bfrag(w_enc, CS, n, 32 + kb));
#pragma unroll
  for (int t = 0; t < 4; ++t) {
    const int col = t * 16 + n;
    put(F_BG + 2 * t,     load_bfrag(w_g1, HID, col, kb));
    put(F_BG + 2 * t + 1, load_bfrag(w_g1, HID, col, 32 + kb));
    put(F_BC + 2 * t,     load_bfrag(w_c1, 48, col, kb));          // K 0..31
    put(F_BC + 2 * t + 1,                                          // K 32..47 | 0-pad
        [&] { v16h v = zero16h(); const float* p = w_c1 + col * 48 + 32 + kb;
              for (int i = 0; i < 8; ++i) v[i] = (_Float16)p[i]; return v; }());
    put(F_BD + 2 * t,     load_bfrag(w_c2, HID, col, kb));
    put(F_BD + 2 * t + 1, load_bfrag(w_c2, HID, col, 32 + kb));
  }
}

// ---------------- main kernel ----------------

__global__ __launch_bounds__(WPB * 32) void dual_head_mutator_kernel(
    const float* __restrict__ A, const float* __restrict__ B,
    const int* __restrict__ coords, const float* __restrict__ noise,
    const float* __restrict__ b_enc, const float* __restrict__ b_c1,
    const float* __restrict__ b_c2, const float* __restrict__ b_g1,
    const float* __restrict__ w_g2, const float* __restrict__ b_g2,
    const float* __restrict__ ref_policy, const float* __restrict__ ref_meta,
    const float* __restrict__ cs_policy, const float* __restrict__ cs_meta,
    const float* __restrict__ es_policy, const float* __restrict__ es_meta,
    const _Float16* __restrict__ wsW, float* __restrict__ out) {
  __shared__ float lds0[WPB][16 * LSTR];   // transient: gate_in -> x -> hidden2
  __shared__ float lds1[WPB][16 * LSTR];   // base tile (f32)
  __shared__ float lalpha[WPB][16];

  const int lane = threadIdx.x & 31;
  const int wv   = threadIdx.x >> 5;
  const int tile = blockIdx.x * WPB + wv;
  const int c0 = tile * 16;                // first chunk of tile
  const int s  = lane >> 4;                // lane half (A/B layout K-offset)
  const int n  = lane & 15;                // column (C layout) / row (A layout)

  float* buf0 = lds0[wv];
  float* buf1 = lds1[wv];

  const float csP = *cs_policy, csM = *cs_meta;
  const float esP = *es_policy, esM = *es_meta;
  const int   split = coords[0];
  const float rp = ref_policy[n], rm = ref_meta[n];
  const float be = b_enc[n];

  // ---- load a/b chunk tiles in A-fragment layout (keep f32 for the blend) ----
  float a0[16], a1[16], bb0[16], bb1[16];
  {
    const float* ar = A + (size_t)(c0 + n) * CS;
    const float* br = B + (size_t)(c0 + n) * CS;
    read_octets_nt(ar + s * 8,      a0);
    read_octets_nt(ar + 32 + s * 8, a1);
    read_octets_nt(br + s * 8,      bb0);
    read_octets_nt(br + 32 + s * 8, bb1);
  }
  v16h aF0 = tohalf(a0), aF1 = tohalf(a1);
  v16h bF0 = tohalf(bb0), bF1 = tohalf(bb1);

  const v16h We0 = load_wsfrag(wsW, F_WE0, lane);
  const v16h We1 = load_wsfrag(wsW, F_WE1, lane);

  // ---- encode a and b: ea/eb = tanh(x @ w_enc^T + b_enc); bias rides in C ----
  const v8f cbe = bias8(be);
  v8f ca = cbe; ca = wmma_f16(aF0, We0, ca); ca = wmma_f16(aF1, We1, ca);
  v8f cb = cbe; cb = wmma_f16(bF0, We0, cb); cb = wmma_f16(bF1, We1, cb);

  float ea[8], eb[8];
#pragma unroll
  for (int r = 0; r < 8; ++r) {
    ea[r] = fast_tanh(ca[r]);
    eb[r] = fast_tanh(cb[r]);
  }

  // ---- build gate_in = [ea | eb | ea-eb | ea*eb] (16x64) in LDS ----
#pragma unroll
  for (int r = 0; r < 8; ++r) {
    float* rowp = buf0 + (r + s * 8) * LSTR;
    rowp[n]      = ea[r];
    rowp[16 + n] = eb[r];
    rowp[32 + n] = ea[r] - eb[r];
    rowp[48 + n] = ea[r] * eb[r];
  }

  // ---- gate_in back as A-fragments ----
  float t0[16], t1[16];
  read_octets(buf0 + n * LSTR + s * 8, t0);
  read_octets(buf0 + n * LSTR + 32 + s * 8, t1);
  v16h gF0 = tohalf(t0), gF1 = tohalf(t1);

  // ---- gate MLP: h = tanh(gate_in @ w_g1^T + b_g1); layer2 (N=2) folded into
  //      a shuffle reduction against (w_g2[0]-w_g2[1]) ----
  float hred[8];
#pragma unroll
  for (int r = 0; r < 8; ++r) hred[r] = 0.0f;
#pragma unroll
  for (int t = 0; t < 4; ++t) {
    const int col = t * 16 + n;
    v16h Bg0 = load_wsfrag(wsW, F_BG + 2 * t, lane);
    v16h Bg1 = load_wsfrag(wsW, F_BG + 2 * t + 1, lane);
    v8f c = bias8(b_g1[col]);
    c = wmma_f16(gF0, Bg0, c); c = wmma_f16(gF1, Bg1, c);
    const float wd = w_g2[col] - w_g2[HID + col];
#pragma unroll
    for (int r = 0; r < 8; ++r) hred[r] += fast_tanh(c[r]) * wd;
  }
#pragma unroll
  for (int m = 1; m < 16; m <<= 1) {
#pragma unroll
    for (int r = 0; r < 8; ++r) hred[r] += __shfl_xor(hred[r], m, 32);
  }
  const float bgd = b_g2[0] - b_g2[1];
  if (n == 0) {
#pragma unroll
    for (int r = 0; r < 8; ++r)
      lalpha[wv][s * 8 + r] = fast_sigmoid(hred[r] + bgd);
  }
  const float alpha = lalpha[wv][n];   // alpha for this lane's A-layout row

  // ---- base = alpha*a + (1-alpha)*b; stash f32 tile in LDS, f16 A-frags ----
  float s0[16], s1[16];
#pragma unroll
  for (int i = 0; i < 16; ++i) {
    s0[i] = bb0[i] + alpha * (a0[i] - bb0[i]);
    s1[i] = bb1[i] + alpha * (a1[i] - bb1[i]);
  }
  write_octets(buf1 + n * LSTR + s * 8, s0);
  write_octets(buf1 + n * LSTR + 32 + s * 8, s1);
  v16h sF0 = tohalf(s0), sF1 = tohalf(s1);

  // ---- encode base; build x = [enc | ref | enc-ref | 0] (16x64, K padded) ----
  v8f ce = cbe; ce = wmma_f16(sF0, We0, ce); ce = wmma_f16(sF1, We1, ce);
#pragma unroll
  for (int r = 0; r < 8; ++r) {
    const int m = r + s * 8;
    const bool pol = (c0 + m) * CS < split;
    const float rv = pol ? rp : rm;
    const float e  = fast_tanh(ce[r]);
    float* rowp = buf0 + m * LSTR;
    rowp[n]      = e;
    rowp[16 + n] = rv;
    rowp[32 + n] = e - rv;
    rowp[48 + n] = 0.0f;
  }

  // ---- x as A-fragments ----
  read_octets(buf0 + n * LSTR + s * 8, t0);
  read_octets(buf0 + n * LSTR + 32 + s * 8, t1);
  v16h xF0 = tohalf(t0), xF1 = tohalf(t1);

  // ---- corrector layer 1: hidden2 = tanh(x @ w_c1^T + b_c1)  (K=48 padded) ----
#pragma unroll
  for (int t = 0; t < 4; ++t) {
    const int col = t * 16 + n;
    v16h Bc0 = load_wsfrag(wsW, F_BC + 2 * t, lane);
    v16h Bc1 = load_wsfrag(wsW, F_BC + 2 * t + 1, lane);   // pre-zero-padded
    v8f c = bias8(b_c1[col]);
    c = wmma_f16(xF0, Bc0, c); c = wmma_f16(xF1, Bc1, c);
#pragma unroll
    for (int r = 0; r < 8; ++r)
      buf0[(r + s * 8) * LSTR + t * 16 + n] = fast_tanh(c[r]);
  }

  // ---- hidden2 as A-fragments ----
  read_octets(buf0 + n * LSTR + s * 8, t0);
  read_octets(buf0 + n * LSTR + 32 + s * 8, t1);
  v16h hF0 = tohalf(t0), hF1 = tohalf(t1);

  // ---- corrector layer 2 + clip + noise + output (non-temporal streams) ----
#pragma unroll
  for (int t = 0; t < 4; ++t) {
    const int col = t * 16 + n;
    v16h Bd0 = load_wsfrag(wsW, F_BD + 2 * t, lane);
    v16h Bd1 = load_wsfrag(wsW, F_BD + 2 * t + 1, lane);
    v8f c = bias8(b_c2[col]);
    c = wmma_f16(hF0, Bd0, c); c = wmma_f16(hF1, Bd1, c);
#pragma unroll
    for (int r = 0; r < 8; ++r) {
      const int m = r + s * 8;
      const int chunk = c0 + m;
      const bool pol = chunk * CS < split;
      float corr = fast_tanh(c[r]) * (pol ? csP : csM);
      corr = fminf(0.1f, fmaxf(-0.1f, corr));
      const float bas = buf1[m * LSTR + t * 16 + n];
      const size_t idx = (size_t)chunk * CS + t * 16 + n;
      const float nz = __builtin_nontemporal_load(noise + idx);
      __builtin_nontemporal_store(bas + corr + nz * (pol ? esP : esM), out + idx);
    }
  }
}

extern "C" void kernel_launch(void* const* d_in, const int* in_sizes, int n_in,
                              void* d_out, int out_size, void* d_ws, size_t ws_size,
                              hipStream_t stream) {
  (void)in_sizes; (void)n_in; (void)out_size; (void)ws_size;
  const float* A      = (const float*)d_in[0];
  const float* B      = (const float*)d_in[1];
  const int*   coords = (const int*)d_in[2];
  const float* noise  = (const float*)d_in[3];
  const float* w_enc  = (const float*)d_in[4];
  const float* b_enc  = (const float*)d_in[5];
  const float* w_c1   = (const float*)d_in[6];
  const float* b_c1   = (const float*)d_in[7];
  const float* w_c2   = (const float*)d_in[8];
  const float* b_c2   = (const float*)d_in[9];
  const float* w_g1   = (const float*)d_in[10];
  const float* b_g1   = (const float*)d_in[11];
  const float* w_g2   = (const float*)d_in[12];
  const float* b_g2   = (const float*)d_in[13];
  const float* refP   = (const float*)d_in[14];
  const float* refM   = (const float*)d_in[15];
  const float* csP    = (const float*)d_in[16];
  const float* csM    = (const float*)d_in[17];
  const float* esP    = (const float*)d_in[18];
  const float* esM    = (const float*)d_in[19];
  float* out = (float*)d_out;
  _Float16* wsW = (_Float16*)d_ws;   // needs only 26 KB of workspace

  // Stage 1: pack weights into f16 WMMA B-fragment layout (single wave).
  hipLaunchKernelGGL(prepack_weights_kernel, dim3(1), dim3(32), 0, stream,
                     w_enc, w_g1, w_c1, w_c2, wsW);

  // Stage 2: fused gated-crossover + dual-head correction + noise.
  dim3 grid(TILES / WPB);     // 8192 blocks
  dim3 block(WPB * 32);       // 4 wave32s
  hipLaunchKernelGGL(dual_head_mutator_kernel, grid, block, 0, stream,
                     A, B, coords, noise, b_enc, b_c1, b_c2, b_g1, w_g2, b_g2,
                     refP, refM, csP, csM, esP, esM, wsW, out);
}
